// GlobalCrossAttention_6433861009696
// MI455X (gfx1250) — matrix operand order repaired and checked by reading, not compiled, not run on hardware
//
#include <hip/hip_runtime.h>
#include <hip/hip_bf16.h>

typedef __attribute__((ext_vector_type(16))) _Float16 v16h;
typedef __attribute__((ext_vector_type(8)))  float    v8f;

#define DIM_C 256
#define HEADS_C 8
#define HD_C 32
#define B_C 4
#define NQ_C 300
#define N_C 4096
#define GRID_C 64   // HH == WW == 64
#define RPEH_C 512

__device__ __forceinline__ v8f wmma_f16(v16h a, v16h b, v8f c) {
  // D = A(16x32 f16) * B(32x16 f16) + C(16x16 f32)
  return __builtin_amdgcn_wmma_f32_16x16x32_f16(
      /*neg_a=*/false, a, /*neg_b=*/false, b,
      /*c_mod=*/(short)0, c, /*reuse_a=*/false, /*reuse_b=*/false);
}

// ---------------------------------------------------------------------------
// Projection GEMM: out = (X[M x 256] @ W[256 x 256] + bias) * scale
// One wave per 16x16 output tile; K=256 in 8 chunks of 32 via WMMA f16.
// SEQ/MODE are compile-time so the m/seq, m%seq epilogue lowers to
// magic-number multiplies (SEQ=300) or shifts (SEQ=4096) instead of the
// ~25-VALU software division sequence.
// MODE 0: q -> head layout [B][8][300][32]
// MODE 1: k -> head layout [B][8][4096][32]
// MODE 2: v -> transposed   [B][8][32][4096]  (so PV B-frags are contiguous)
// W tiles are reused by every M-tile, so the strided per-lane W loads hit
// the 192 MB L2 after first touch; no LDS staging needed.
// ---------------------------------------------------------------------------
template <int SEQ, int MODE>
__global__ __launch_bounds__(32) void proj_kernel(
    const float* __restrict__ X, const float* __restrict__ W,
    const float* __restrict__ bias, _Float16* __restrict__ out, float scale) {
  const int lane = threadIdx.x;
  const int c = lane & 15, hi = lane >> 4;
  const int m0 = blockIdx.x * 16, n0 = blockIdx.y * 16;

  v8f acc = {};
  const float* xr = X + (size_t)(m0 + c) * DIM_C;
  for (int kc = 0; kc < DIM_C; kc += 32) {
    v16h a, bf;
#pragma unroll
    for (int i = 0; i < 8; ++i) a[i] = (_Float16)xr[kc + hi * 16 + i];
#pragma unroll
    for (int i = 0; i < 8; ++i) a[8 + i] = (_Float16)xr[kc + hi * 16 + 8 + i];
#pragma unroll
    for (int e = 0; e < 16; ++e)
      bf[e] = (_Float16)W[(size_t)(kc + hi * 16 + e) * DIM_C + n0 + c];
    acc = wmma_f16(a, bf, acc);
  }

  const int n = n0 + c;
  const float bv = bias[n];
  const int head = n >> 5, d = n & 31;
#pragma unroll
  for (int j = 0; j < 8; ++j) {
    const int m = m0 + j + 8 * hi;
    const int bidx = m / SEQ, s = m % SEQ;  // compile-time SEQ -> no div
    const float val = (acc[j] + bv) * scale;
    if (MODE == 2) {
      out[(((size_t)bidx * HEADS_C + head) * HD_C + d) * (size_t)SEQ + s] =
          (_Float16)val;
    } else {
      out[(((size_t)bidx * HEADS_C + head) * (size_t)SEQ + s) * HD_C + d] =
          (_Float16)val;
    }
  }
}

// ---------------------------------------------------------------------------
// RPE MLP: per (b,q,pos,axis): relu([d0,d1] @ W1 + b1) @ W2 -> 8 heads.
// Weights staged in LDS (22.5 KB of 320 KB WGP pool). Pure VALU (K=2, N=8
// is far too skinny for WMMA). Writes rpe[axis][b][q][pos][8] f32.
// ---------------------------------------------------------------------------
__global__ __launch_bounds__(256) void rpe_kernel(
    const float* __restrict__ refpts,  // [B][NQ][1][4] = cx,cy,w,h
    const float* __restrict__ W1x, const float* __restrict__ b1x,
    const float* __restrict__ W2x, const float* __restrict__ W1y,
    const float* __restrict__ b1y, const float* __restrict__ W2y,
    float* __restrict__ rpex, float* __restrict__ rpey) {
  const int axis = blockIdx.y;
  const float* W1 = axis ? W1y : W1x;
  const float* b1 = axis ? b1y : b1x;
  const float* W2 = axis ? W2y : W2x;
  float* outp = axis ? rpey : rpex;

  __shared__ float sW1a[RPEH_C], sW1b[RPEH_C], sb1[RPEH_C];
  __shared__ float sW2[RPEH_C * HEADS_C];
  const int tid = threadIdx.x;
  for (int i = tid; i < RPEH_C; i += 256) {
    sW1a[i] = W1[i];            // W1[0][i]
    sW1b[i] = W1[RPEH_C + i];   // W1[1][i]
    sb1[i] = b1[i];
  }
  for (int i = tid; i < RPEH_C * HEADS_C; i += 256) sW2[i] = W2[i];
  __syncthreads();

  const int pos = blockIdx.x * 256 + tid;  // < B*NQ*64
  if (pos >= B_C * NQ_C * GRID_C) return;
  const int p = pos & 63;
  const int bq = pos >> 6;
  const float* rp = refpts + (size_t)bq * 4;
  const float ctr = rp[axis];
  const float sz = rp[2 + axis];
  const float pv = (p + 0.5f) * 16.0f;  // STRIDE=16
  const float d0 = ctr - 0.5f * sz - pv;
  const float d1 = ctr + 0.5f * sz - pv;

  float acc[HEADS_C] = {0.f, 0.f, 0.f, 0.f, 0.f, 0.f, 0.f, 0.f};
  for (int j = 0; j < RPEH_C; ++j) {
    float h = fmaf(d0, sW1a[j], fmaf(d1, sW1b[j], sb1[j]));
    h = fmaxf(h, 0.f);
#pragma unroll
    for (int k = 0; k < HEADS_C; ++k)
      acc[k] = fmaf(h, sW2[j * HEADS_C + k], acc[k]);
  }
  float* o = outp + (size_t)pos * HEADS_C;
#pragma unroll
  for (int k = 0; k < HEADS_C; ++k) o[k] = acc[k];
}

// ---------------------------------------------------------------------------
// Fused flash attention with decomposed RPE + padding mask.
// One wave per (b, h, 16-query tile). Streams 4096 keys in 32-key chunks:
//   2x QK wmma (K=HD=32 contraction) + online softmax + 2x PV wmma.
// Never materializes the (B,H,NQ,N) logits/rpe tensors (saves ~470 MB HBM).
// K/V chunks are prefetched one iteration ahead (global_prefetch_b8) to
// overlap L2 latency with the softmax VALU work.
// ---------------------------------------------------------------------------
__global__ __launch_bounds__(32) void attn_kernel(
    const _Float16* __restrict__ qh,   // [B][8][300][32] (pre-scaled)
    const _Float16* __restrict__ kh,   // [B][8][4096][32]
    const _Float16* __restrict__ vt,   // [B][8][32][4096]
    const float* __restrict__ rpex,    // [B][300][64][8]
    const float* __restrict__ rpey,    // [B][300][64][8]
    const int* __restrict__ mask,      // [B][4096]
    _Float16* __restrict__ xh) {       // [B][300][256]
  const int q0 = blockIdx.x * 16;
  const int h = blockIdx.y;
  const int b = blockIdx.z;
  const int lane = threadIdx.x;
  const int c = lane & 15, hi = lane >> 4;

  __shared__ float rx[16][GRID_C];
  __shared__ float ry[16][GRID_C];
  __shared__ _Float16 Ps[16][32];

  // Stage the two rpe vectors for this q-tile / head (16 x 64 each).
  for (int idx = lane; idx < 16 * GRID_C; idx += 32) {
    const int r = idx >> 6, w = idx & 63;
    const int q = q0 + r;
    float vx = 0.f, vy = 0.f;
    if (q < NQ_C) {
      const size_t o = (((size_t)b * NQ_C + q) * GRID_C + w) * HEADS_C + h;
      vx = rpex[o];
      vy = rpey[o];
    }
    rx[r][w] = vx;
    ry[r][w] = vy;
  }
  __syncthreads();

  // Q fragment (A layout: row = lane&15; K halves split by lane>>4).
  const _Float16* qbase = qh + ((size_t)b * HEADS_C + h) * NQ_C * HD_C;
  v16h aq;
#pragma unroll
  for (int i = 0; i < 16; ++i) aq[i] = (_Float16)0.f;
  if (q0 + c < NQ_C) {
    const _Float16* row = qbase + (size_t)(q0 + c) * HD_C;
#pragma unroll
    for (int i = 0; i < 8; ++i) aq[i] = row[hi * 16 + i];
#pragma unroll
    for (int i = 0; i < 8; ++i) aq[8 + i] = row[hi * 16 + 8 + i];
  }

  const _Float16* kbase = kh + ((size_t)b * HEADS_C + h) * (size_t)N_C * HD_C;
  const _Float16* vbase = vt + ((size_t)b * HEADS_C + h) * (size_t)HD_C * N_C;
  const int* mbase = mask + (size_t)b * N_C;

  float mrow[8], lrow[8];
#pragma unroll
  for (int j = 0; j < 8; ++j) { mrow[j] = -1e30f; lrow[j] = 0.f; }
  v8f o0 = {};  // dims 0..15
  v8f o1 = {};  // dims 16..31

  for (int kc = 0; kc < N_C; kc += 32) {
    // Prefetch next chunk's K rows and V segments (global_prefetch_b8).
    if (kc + 32 < N_C) {
      __builtin_prefetch(kbase + (size_t)(kc + 32 + lane) * HD_C, 0, 1);
      __builtin_prefetch(vbase + (size_t)lane * N_C + kc + 32, 0, 1);
    }

    // K^T B-fragments: lane c = key column, hi = dim half (contiguous 32B).
    v16h bk0, bk1;
    {
      const _Float16* k0 = kbase + (size_t)(kc + c) * HD_C + hi * 16;
      const _Float16* k1 = kbase + (size_t)(kc + 16 + c) * HD_C + hi * 16;
#pragma unroll
      for (int e = 0; e < 16; ++e) { bk0[e] = k0[e]; bk1[e] = k1[e]; }
    }
    v8f z = {};
    v8f s0 = wmma_f16(aq, bk0, z);
    v8f s1 = wmma_f16(aq, bk1, z);

    const float mb0 = -100.f * (float)mbase[kc + c];
    const float mb1 = -100.f * (float)mbase[kc + 16 + c];
    const int hh = kc >> 6;      // key row (constant across the 32-key chunk)
    const int ww0 = kc & 63;     // 0 or 32

    float p0[8], p1[8];
#pragma unroll
    for (int j = 0; j < 8; ++j) {
      const int r = j + 8 * hi;
      const float ryv = ry[r][hh];
      float t0 = s0[j] + rx[r][ww0 + c] + ryv + mb0;
      float t1 = s1[j] + rx[r][ww0 + 16 + c] + ryv + mb1;
      float mx = fmaxf(t0, t1);
      for (int off = 8; off; off >>= 1) mx = fmaxf(mx, __shfl_xor(mx, off, 16));
      const float mnew = fmaxf(mrow[j], mx);
      const float scl = __expf(mrow[j] - mnew);
      p0[j] = __expf(t0 - mnew);
      p1[j] = __expf(t1 - mnew);
      float ls = p0[j] + p1[j];
      for (int off = 8; off; off >>= 1) ls += __shfl_xor(ls, off, 16);
      lrow[j] = lrow[j] * scl + ls;
      mrow[j] = mnew;
      o0[j] *= scl;
      o1[j] *= scl;
    }

    // Bounce P through LDS to convert C-layout -> A-layout.
    __syncthreads();
#pragma unroll
    for (int j = 0; j < 8; ++j) {
      const int r = j + 8 * hi;
      Ps[r][c] = (_Float16)p0[j];
      Ps[r][16 + c] = (_Float16)p1[j];
    }
    __syncthreads();
    v16h ap;
#pragma unroll
    for (int i = 0; i < 8; ++i) ap[i] = Ps[c][hi * 16 + i];
#pragma unroll
    for (int i = 0; i < 8; ++i) ap[8 + i] = Ps[c][hi * 16 + 8 + i];

    // V B-fragments from pre-transposed layout: contiguous 32B per lane.
    v16h bv0, bv1;
    {
      const _Float16* v0 = vbase + (size_t)c * N_C + kc + hi * 16;
      const _Float16* v1 = vbase + (size_t)(16 + c) * N_C + kc + hi * 16;
#pragma unroll
      for (int e = 0; e < 16; ++e) { bv0[e] = v0[e]; bv1[e] = v1[e]; }
    }
    o0 = wmma_f16(ap, bv0, o0);
    o1 = wmma_f16(ap, bv1, o1);
  }

  // Epilogue: normalize and emit f16 rows for the output projection.
#pragma unroll
  for (int j = 0; j < 8; ++j) {
    const int q = q0 + j + 8 * hi;
    if (q < NQ_C) {
      const float inv = 1.f / lrow[j];
      const size_t base = ((size_t)b * NQ_C + q) * DIM_C + h * HD_C;
      xh[base + c] = (_Float16)(o0[j] * inv);
      xh[base + 16 + c] = (_Float16)(o1[j] * inv);
    }
  }
}

// ---------------------------------------------------------------------------
// Output projection: d_out = Xh[1200 x 256](f16) @ Wp + bp  (f32 out)
// ---------------------------------------------------------------------------
__global__ __launch_bounds__(32) void outproj_kernel(
    const _Float16* __restrict__ Xh, const float* __restrict__ W,
    const float* __restrict__ bias, float* __restrict__ out) {
  const int lane = threadIdx.x;
  const int c = lane & 15, hi = lane >> 4;
  const int m0 = blockIdx.x * 16, n0 = blockIdx.y * 16;

  v8f acc = {};
  const _Float16* xr = Xh + (size_t)(m0 + c) * DIM_C;
  for (int kc = 0; kc < DIM_C; kc += 32) {
    v16h a, bf;
#pragma unroll
    for (int i = 0; i < 8; ++i) a[i] = xr[kc + hi * 16 + i];
#pragma unroll
    for (int i = 0; i < 8; ++i) a[8 + i] = xr[kc + hi * 16 + 8 + i];
#pragma unroll
    for (int e = 0; e < 16; ++e)
      bf[e] = (_Float16)W[(size_t)(kc + hi * 16 + e) * DIM_C + n0 + c];
    acc = wmma_f16(a, bf, acc);
  }
  const int n = n0 + c;
  const float bv = bias[n];
#pragma unroll
  for (int j = 0; j < 8; ++j) {
    const int m = m0 + j + 8 * hi;
    out[(size_t)m * DIM_C + n] = acc[j] + bv;
  }
}

// ---------------------------------------------------------------------------
extern "C" void kernel_launch(void* const* d_in, const int* in_sizes, int n_in,
                              void* d_out, int out_size, void* d_ws,
                              size_t ws_size, hipStream_t stream) {
  const float* query = (const float*)d_in[0];
  const float* refpts = (const float*)d_in[1];
  const float* k_in = (const float*)d_in[2];
  const float* v_in = (const float*)d_in[3];
  // d_in[4] = input_spatial_shapes (statically 64x64)
  const int* pad_mask = (const int*)d_in[5];
  const float* Wq = (const float*)d_in[6];
  const float* bq = (const float*)d_in[7];
  const float* Wk = (const float*)d_in[8];
  const float* bk = (const float*)d_in[9];
  const float* Wv = (const float*)d_in[10];
  const float* bv = (const float*)d_in[11];
  const float* Wp = (const float*)d_in[12];
  const float* bp = (const float*)d_in[13];
  const float* W1x = (const float*)d_in[14];
  const float* b1x = (const float*)d_in[15];
  const float* W2x = (const float*)d_in[16];
  const float* W1y = (const float*)d_in[17];
  const float* b1y = (const float*)d_in[18];
  const float* W2y = (const float*)d_in[19];

  // Workspace carve-up (all 256B aligned), ~22 MB total.
  char* ws = (char*)d_ws;
  size_t off = 0;
  _Float16* qh = (_Float16*)(ws + off); off += (size_t)B_C * HEADS_C * NQ_C * HD_C * 2;           // 614400
  _Float16* kh = (_Float16*)(ws + off); off += (size_t)B_C * HEADS_C * N_C * HD_C * 2;            // 8 MB
  _Float16* vt = (_Float16*)(ws + off); off += (size_t)B_C * HEADS_C * N_C * HD_C * 2;            // 8 MB
  float* rpex = (float*)(ws + off); off += (size_t)B_C * NQ_C * GRID_C * HEADS_C * 4;             // 2.4 MB
  float* rpey = (float*)(ws + off); off += (size_t)B_C * NQ_C * GRID_C * HEADS_C * 4;             // 2.4 MB
  _Float16* xh = (_Float16*)(ws + off); off += (size_t)B_C * NQ_C * DIM_C * 2;                    // 614400
  (void)ws_size;

  const float qscale = 0.17677669529663689f;  // 32^-0.5

  // Projections (f16 WMMA, f32 accumulate).
  proj_kernel<NQ_C, 0><<<dim3((B_C * NQ_C) / 16, DIM_C / 16), 32, 0, stream>>>(
      query, Wq, bq, qh, qscale);
  proj_kernel<N_C, 1><<<dim3((B_C * N_C) / 16, DIM_C / 16), 32, 0, stream>>>(
      k_in, Wk, bk, kh, 1.f);
  proj_kernel<N_C, 2><<<dim3((B_C * N_C) / 16, DIM_C / 16), 32, 0, stream>>>(
      v_in, Wv, bv, vt, 1.f);

  // Decomposed RPE MLPs (x and y axes).
  rpe_kernel<<<dim3((B_C * NQ_C * GRID_C + 255) / 256, 2), 256, 0, stream>>>(
      refpts, W1x, b1x, W2x, W1y, b1y, W2y, rpex, rpey);

  // Fused flash attention.
  attn_kernel<<<dim3((NQ_C + 15) / 16, HEADS_C, B_C), 32, 0, stream>>>(
      qh, kh, vt, rpex, rpey, pad_mask, xh);

  // Output projection.
  outproj_kernel<<<dim3((B_C * NQ_C) / 16, DIM_C / 16), 32, 0, stream>>>(
      xh, Wp, bp, (float*)d_out);
}
